// MultiHeadAttention_60885456388587
// MI455X (gfx1250) — matrix operand order repaired and checked by reading, not compiled, not run on hardware
//
#include <hip/hip_runtime.h>
#include <hip/hip_bf16.h>

// ---------------------------------------------------------------------------
// Multi-head attention forward for MI455X (gfx1250), wave32 + WMMA bf16.
//   qkv = x @ W_qkv + b_qkv       (bf16 WMMA, f32 accum)
//   attn = flash-attention        (bf16 WMMA, f32 accum, online softmax)
//   out  = attn @ W_proj + b_proj (bf16 WMMA, f32 accum, f32 out)
// ---------------------------------------------------------------------------

typedef __bf16 bf16;
typedef __attribute__((ext_vector_type(16))) __bf16 bf16x16;
typedef __attribute__((ext_vector_type(8)))  __bf16 bf16x8;
typedef __attribute__((ext_vector_type(8)))  float  f32x8;

union Bf16Frag { bf16x16 v; bf16x8 h[2]; };

__device__ __forceinline__ unsigned short bf_bits(float f) {
  union { bf16 b; unsigned short s; } u;
  u.b = (bf16)f;
  return u.s;
}

// A-matrix fragment (16x32 bf16), per CDNA5 ISA 7.12.2:
// lane = row (0..15); VGPR0..3 hold K = hi*8..+7, VGPR4..7 hold K = 16+hi*8..+7.
__device__ __forceinline__ bf16x16 load_frag_a(const bf16* base, int ldk, int kbase) {
  const int lane = threadIdx.x & 31;
  const int row = lane & 15, hi = lane >> 4;
  const bf16* p = base + row * ldk + kbase + hi * 8;
  Bf16Frag f;
  f.h[0] = *(const bf16x8*)(p);
  f.h[1] = *(const bf16x8*)(p + 16);
  return f.v;
}

// B-matrix fragment (32x16 bf16): lane = column N; elems 0..7 hold K = hi*8..+7,
// elems 8..15 hold K = 16+hi*8..+7 (matches A). Source tile stored as [N][K].
__device__ __forceinline__ bf16x16 load_frag_b(const bf16* baseT, int ldk, int kbase) {
  const int lane = threadIdx.x & 31;
  const int col = lane & 15, hi = lane >> 4;
  const bf16* p = baseT + col * ldk + kbase + hi * 8;
  Bf16Frag f;
  f.h[0] = *(const bf16x8*)(p);
  f.h[1] = *(const bf16x8*)(p + 16);
  return f.v;
}

#define WMMA_BF16(A, B, C) \
  __builtin_amdgcn_wmma_f32_16x16x32_bf16(false, (A), false, (B), (short)0, (C), false, false)

__device__ __forceinline__ void lds_fence() {
  asm volatile("s_wait_dscnt 0" ::: "memory");
}

// ---------------------------------------------------------------------------
// f32 -> bf16 conversion (vectorized), n must be a multiple of 4
// ---------------------------------------------------------------------------
__global__ __launch_bounds__(256) void f32_to_bf16_kernel(
    const float* __restrict__ in, bf16* __restrict__ out, int n) {
  int i = (blockIdx.x * 256 + threadIdx.x) * 4;
  if (i + 3 < n) {
    float4 v = *(const float4*)(in + i);
    out[i + 0] = (bf16)v.x;
    out[i + 1] = (bf16)v.y;
    out[i + 2] = (bf16)v.z;
    out[i + 3] = (bf16)v.w;
  }
}

// ---------------------------------------------------------------------------
// Tiled bf16 GEMM: C[M,N] = A[M,K] @ B[K,N] + bias[N]
// Block tile 128x128, K-tile 64, 8 waves (each wave: 32x64 -> 8 f32x8 accums)
// ---------------------------------------------------------------------------
__global__ __launch_bounds__(256) void gemm_bf16_kernel(
    const bf16* __restrict__ A, const bf16* __restrict__ B,
    const float* __restrict__ bias, void* __restrict__ Cout,
    int M, int N, int K, int out_is_bf16) {
  __shared__ __attribute__((aligned(16))) bf16 As[128 * 64];
  __shared__ __attribute__((aligned(16))) bf16 Bts[128 * 64];   // [N][K]

  const int tid = threadIdx.x;
  const int lane = tid & 31, w = tid >> 5;
  const int wy = w & 3, wx = w >> 2;            // 4x2 wave grid
  const int hi = lane >> 4, ln = lane & 15;
  const int m0 = blockIdx.y * 128, n0 = blockIdx.x * 128;

  f32x8 acc[2][4] = {};

  for (int k0 = 0; k0 < K; k0 += 64) {
    __syncthreads();

    // ---- Load phase: issue all global loads first (overlap in LOADcnt) ----
    // A tile: 128 rows x 64 cols, 16B chunks, 4 per thread
    uint4 areg[4];
#pragma unroll
    for (int i = 0; i < 4; ++i) {
      int c = tid + i * 256;
      int row = c >> 3, col = (c & 7) * 8;
      areg[i] = *(const uint4*)&A[(size_t)(m0 + row) * K + k0 + col];
    }
    // B tile: each thread loads K rows (2kp, 2kp+1) x 16 columns
    const int kp = tid >> 3;            // 0..31 -> k = 2*kp
    const int nseg = tid & 7;           // 0..7  -> 16 n each
    const bf16* src0 = &B[(size_t)(k0 + 2 * kp) * N + n0 + nseg * 16];
    union { uint4 q[2]; unsigned short e[16]; } u0, u1;
    u0.q[0] = *(const uint4*)(src0);
    u0.q[1] = *(const uint4*)(src0 + 8);
    u1.q[0] = *(const uint4*)(src0 + N);
    u1.q[1] = *(const uint4*)(src0 + N + 8);
    if (k0 + 64 < K)   // L2 prefetch of next A tile (global_prefetch_b8)
      __builtin_prefetch(&A[(size_t)(m0 + (tid >> 1)) * K + k0 + 64], 0, 1);

    // ---- Store phase: LDS writes ----
#pragma unroll
    for (int i = 0; i < 4; ++i) {
      int c = tid + i * 256;
      int row = c >> 3, col = (c & 7) * 8;
      *(uint4*)&As[row * 64 + col] = areg[i];
    }
    // Transposed B: pack K-pairs into dwords -> ds_store_b32
    unsigned int* Bts32 = (unsigned int*)Bts;
#pragma unroll
    for (int j = 0; j < 16; ++j) {
      unsigned int v = (unsigned int)u0.e[j] | ((unsigned int)u1.e[j] << 16);
      Bts32[(nseg * 16 + j) * 32 + kp] = v;
    }
    __syncthreads();

    // ---- Compute: preload fragments, then WMMA chain ----
#pragma unroll
    for (int kk = 0; kk < 64; kk += 32) {
      bf16x16 a0 = load_frag_a(&As[(wy * 32) * 64], 64, kk);
      bf16x16 a1 = load_frag_a(&As[(wy * 32 + 16) * 64], 64, kk);
      bf16x16 bfr[4];
#pragma unroll
      for (int nt = 0; nt < 4; ++nt)
        bfr[nt] = load_frag_b(&Bts[(wx * 64 + nt * 16) * 64], 64, kk);
#pragma unroll
      for (int nt = 0; nt < 4; ++nt) {
        acc[0][nt] = WMMA_BF16(a0, bfr[nt], acc[0][nt]);
        acc[1][nt] = WMMA_BF16(a1, bfr[nt], acc[1][nt]);
      }
    }
  }

  // Epilogue: C layout -> lane holds column (ln), rows m = v + 8*hi
#pragma unroll
  for (int nt = 0; nt < 4; ++nt) {
    int ng = n0 + wx * 64 + nt * 16 + ln;
    float bv = bias[ng];
#pragma unroll
    for (int mt = 0; mt < 2; ++mt) {
#pragma unroll
      for (int v = 0; v < 8; ++v) {
        int mg = m0 + wy * 32 + mt * 16 + v + 8 * hi;
        float val = acc[mt][nt][v] + bv;
        if (out_is_bf16) ((bf16*)Cout)[(size_t)mg * N + ng] = (bf16)val;
        else             ((float*)Cout)[(size_t)mg * N + ng] = val;
      }
    }
  }
}

// ---------------------------------------------------------------------------
// Flash attention: grid (T/128, H, B); block = 256 threads = 8 waves.
// Wave w owns 16 q rows; streams 64-key K/V tiles. qkv layout [B*T, 3072]
// with Q at col 0, K at 1024, V at 2048 (head h -> cols h*64..h*64+63).
// Scale 1/sqrt(64)=0.125 folded into Q staging.
// ---------------------------------------------------------------------------
__global__ __launch_bounds__(256) void attn_kernel(
    const bf16* __restrict__ qkv, bf16* __restrict__ out) {
  __shared__ __attribute__((aligned(16))) bf16 Qs[128 * 64];
  __shared__ __attribute__((aligned(16))) bf16 Ks[64 * 64];     // [key][d]
  __shared__ __attribute__((aligned(16))) bf16 Vt[64 * 64];     // [d][key]
  __shared__ __attribute__((aligned(16))) bf16 Ps[8 * 16 * 64]; // per-wave P

  const int tid = threadIdx.x, lane = tid & 31, w = tid >> 5;
  const int hi = lane >> 4, ln = lane & 15;
  const int qb = blockIdx.x, h = blockIdx.y, b = blockIdx.z;
  const int T = 2048, C3 = 3072;
  const size_t baseRow = (size_t)b * T;

  // ---- Stage Q (scaled by 0.125), packed b32 LDS stores ----
  {
    uint4 qreg[4];
#pragma unroll
    for (int i = 0; i < 4; ++i) {
      int c = tid + i * 256;
      int row = c >> 3, col = (c & 7) * 8;
      qreg[i] = *(const uint4*)&qkv[(baseRow + qb * 128 + row) * C3 + h * 64 + col];
    }
    unsigned int* Qs32 = (unsigned int*)Qs;
#pragma unroll
    for (int i = 0; i < 4; ++i) {
      int c = tid + i * 256;
      int row = c >> 3, col = (c & 7) * 8;
      union { uint4 q; bf16 e[8]; } u; u.q = qreg[i];
#pragma unroll
      for (int j = 0; j < 4; ++j) {
        unsigned int lo = bf_bits((float)u.e[2 * j] * 0.125f);
        unsigned int hi16 = bf_bits((float)u.e[2 * j + 1] * 0.125f);
        Qs32[(row * 64 + col) / 2 + j] = lo | (hi16 << 16);
      }
    }
  }
  __syncthreads();

  bf16x16 qa0 = load_frag_a(&Qs[(w * 16) * 64], 64, 0);
  bf16x16 qa1 = load_frag_a(&Qs[(w * 16) * 64], 64, 32);

  float mrow[8], lrow[8];
#pragma unroll
  for (int v = 0; v < 8; ++v) { mrow[v] = -1e30f; lrow[v] = 0.f; }
  f32x8 o[4] = {};

  for (int kb = 0; kb < 32; ++kb) {
    __syncthreads();

    // ---- Load phase: all global loads for K and V tiles ----
    uint4 kreg[2];
#pragma unroll
    for (int i = 0; i < 2; ++i) {
      int c = tid + i * 256;
      int row = c >> 3, col = (c & 7) * 8;
      kreg[i] = *(const uint4*)
          &qkv[(baseRow + kb * 64 + row) * C3 + 1024 + h * 64 + col];
    }
    const int kp = tid >> 3;           // key pair 0..31
    const int dseg = tid & 7;          // 8 d values each
    const bf16* vsrc = &qkv[(baseRow + kb * 64 + 2 * kp) * C3 + 2048 + h * 64 + dseg * 8];
    union { uint4 q; unsigned short e[8]; } v0, v1;
    v0.q = *(const uint4*)(vsrc);
    v1.q = *(const uint4*)(vsrc + C3);

    // ---- Store phase ----
#pragma unroll
    for (int i = 0; i < 2; ++i) {
      int c = tid + i * 256;
      int row = c >> 3, col = (c & 7) * 8;
      *(uint4*)&Ks[row * 64 + col] = kreg[i];
    }
    unsigned int* Vt32 = (unsigned int*)Vt;
#pragma unroll
    for (int j = 0; j < 8; ++j) {
      unsigned int pv = (unsigned int)v0.e[j] | ((unsigned int)v1.e[j] << 16);
      Vt32[(dseg * 8 + j) * 32 + kp] = pv;
    }
    __syncthreads();

    // ---- S = Q @ K^T: preload K fragments, then WMMA chain ----
    f32x8 s[4] = {};
    {
      bf16x16 kf[8];
#pragma unroll
      for (int nt = 0; nt < 4; ++nt) {
        kf[2 * nt]     = load_frag_b(&Ks[(nt * 16) * 64], 64, 0);
        kf[2 * nt + 1] = load_frag_b(&Ks[(nt * 16) * 64], 64, 32);
      }
#pragma unroll
      for (int nt = 0; nt < 4; ++nt) {
        s[nt] = WMMA_BF16(qa0, kf[2 * nt], s[nt]);
        s[nt] = WMMA_BF16(qa1, kf[2 * nt + 1], s[nt]);
      }
    }

    // ---- Online softmax (half-wave xor-shuffle row reductions) ----
#pragma unroll
    for (int v = 0; v < 8; ++v) {
      float mloc = fmaxf(fmaxf(s[0][v], s[1][v]), fmaxf(s[2][v], s[3][v]));
#pragma unroll
      for (int off = 1; off < 16; off <<= 1)
        mloc = fmaxf(mloc, __shfl_xor(mloc, off, 32));
      float mn = fmaxf(mrow[v], mloc);
      float alpha = __expf(mrow[v] - mn);
      mrow[v] = mn;
      float psum = 0.f;
#pragma unroll
      for (int nt = 0; nt < 4; ++nt) {
        float p = __expf(s[nt][v] - mn);
        s[nt][v] = p;
        psum += p;
      }
#pragma unroll
      for (int off = 1; off < 16; off <<= 1)
        psum += __shfl_xor(psum, off, 32);
      lrow[v] = lrow[v] * alpha + psum;
#pragma unroll
      for (int dt = 0; dt < 4; ++dt) o[dt][v] *= alpha;
    }

    // ---- P (C layout) -> per-wave LDS so it can re-enter as A fragments ----
#pragma unroll
    for (int nt = 0; nt < 4; ++nt)
#pragma unroll
      for (int v = 0; v < 8; ++v)
        Ps[w * 1024 + (v + 8 * hi) * 64 + nt * 16 + ln] = (bf16)s[nt][v];
    lds_fence();

    // ---- O += P @ V: preload V fragments, then WMMA chain ----
    {
      bf16x16 pa0 = load_frag_a(&Ps[w * 1024], 64, 0);
      bf16x16 pa1 = load_frag_a(&Ps[w * 1024], 64, 32);
      bf16x16 vb[8];
#pragma unroll
      for (int dt = 0; dt < 4; ++dt) {
        vb[2 * dt]     = load_frag_b(&Vt[(dt * 16) * 64], 64, 0);
        vb[2 * dt + 1] = load_frag_b(&Vt[(dt * 16) * 64], 64, 32);
      }
#pragma unroll
      for (int dt = 0; dt < 4; ++dt) {
        o[dt] = WMMA_BF16(pa0, vb[2 * dt], o[dt]);
        o[dt] = WMMA_BF16(pa1, vb[2 * dt + 1], o[dt]);
      }
    }
  }

  // ---- Epilogue: O /= l, write [B*T, 1024] bf16 ----
#pragma unroll
  for (int dt = 0; dt < 4; ++dt) {
#pragma unroll
    for (int v = 0; v < 8; ++v) {
      int m = v + 8 * hi;
      float val = o[dt][v] / lrow[v];
      out[(baseRow + qb * 128 + w * 16 + m) * 1024 + h * 64 + dt * 16 + ln] = (bf16)val;
    }
  }
}

// ---------------------------------------------------------------------------
extern "C" void kernel_launch(void* const* d_in, const int* in_sizes, int n_in,
                              void* d_out, int out_size, void* d_ws, size_t ws_size,
                              hipStream_t stream) {
  const float* x     = (const float*)d_in[0];   // [2,2048,1024]
  const float* Wqkv  = (const float*)d_in[1];   // [1024,3072]
  const float* bqkv  = (const float*)d_in[2];   // [3072]
  const float* Wproj = (const float*)d_in[3];   // [1024,1024]
  const float* bproj = (const float*)d_in[4];   // [1024]
  float* out = (float*)d_out;

  const int Bb = 2, T = 2048, C = 1024;
  const size_t nx   = (size_t)Bb * T * C;       // 4,194,304
  const size_t nwq  = (size_t)C * 3 * C;        // 3,145,728
  const size_t nwp  = (size_t)C * C;            // 1,048,576
  const size_t nqkv = (size_t)Bb * T * 3 * C;   // 12,582,912

  bf16* xb   = (bf16*)d_ws;
  bf16* wqb  = xb + nx;
  bf16* wpb  = wqb + nwq;
  bf16* qkvb = wpb + nwp;
  bf16* attb = qkvb + nqkv;                     // total ~50.3 MB of ws

  f32_to_bf16_kernel<<<(int)(nx  / 1024), 256, 0, stream>>>(x, xb, (int)nx);
  f32_to_bf16_kernel<<<(int)(nwq / 1024), 256, 0, stream>>>(Wqkv, wqb, (int)nwq);
  f32_to_bf16_kernel<<<(int)(nwp / 1024), 256, 0, stream>>>(Wproj, wpb, (int)nwp);

  // qkv = x @ W_qkv + b_qkv  -> bf16 [4096, 3072]
  gemm_bf16_kernel<<<dim3(3072 / 128, 4096 / 128), 256, 0, stream>>>(
      xb, wqb, bqkv, qkvb, 4096, 3072, 1024, 1);

  // flash attention -> bf16 [4096, 1024]
  attn_kernel<<<dim3(T / 128, 16, Bb), 256, 0, stream>>>(qkvb, attb);

  // out = attn @ W_proj + b_proj -> f32 [4096, 1024]
  gemm_bf16_kernel<<<dim3(1024 / 128, 4096 / 128), 256, 0, stream>>>(
      attb, wpb, bproj, out, 4096, 1024, 1024, 0);
}
